// ChildSumTreeLSTM_3152505996048
// MI455X (gfx1250) — matrix operand order repaired and checked by reading, not compiled, not run on hardware
//
#include <hip/hip_runtime.h>
#include <hip/hip_bf16.h>

typedef __attribute__((ext_vector_type(16))) __bf16       v16bf;
typedef __attribute__((ext_vector_type(4)))  __bf16       v4bf;
typedef __attribute__((ext_vector_type(8)))  float        v8f;
typedef __attribute__((ext_vector_type(4)))  unsigned int v4u;
typedef __attribute__((ext_vector_type(2)))  unsigned int v2u;

union Frag { v4u u[2]; v16bf v; };
union H4   { v2u u; v4bf h; };

#define NODES     299593
#define IN_DIM    300
#define MEM       150
#define KPAD_X    320      // IN_DIM padded to 10 K-steps of 32
#define MPAD      160      // MEM padded to 5 K-steps of 32
#define HPAD      160      // padded row stride of hidden-state buffer (16B-aligned rows)
#define CTILES    10
#define NKS_X     10
#define NKS_H     5
#define XW_SZ     (CTILES * NKS_X * 512)   // 51200 bf16 per x-side gate
#define HW_SZ     (CTILES * NKS_H * 512)   // 25600 bf16 per h-side gate
#define LEAF_BASE 37449
#define LEAF_CNT  262144

__device__ __forceinline__ float sigm(float x)   { return 1.0f / (1.0f + __expf(-x)); }
__device__ __forceinline__ float tanh_f(float x) { float e = __expf(2.0f * x); return 1.0f - 2.0f / (e + 1.0f); }

// A fragment (contiguous K order, matching packed B): lane<16 holds K {k0..k0+15},
// lane>=16 holds K {k0+16..k0+31}; row M = lane&15.
__device__ __forceinline__ v16bf load_a(const __bf16* row, int k0, int hi) {
  Frag f;
  f.u[0] = *(const v4u*)(row + k0 + hi * 16);
  f.u[1] = *(const v4u*)(row + k0 + hi * 16 + 8);
  return f.v;
}
// B fragment: pre-swizzled in global memory, 32 contiguous bytes per lane.
__device__ __forceinline__ v16bf load_bg(const __bf16* p) {
  Frag f;
  const v4u* q = (const v4u*)p;
  f.u[0] = q[0];
  f.u[1] = q[1];
  return f.v;
}

// ---------------- Kernel 0: pack weight matrix into fragment-ordered bf16 ----------------
// out[((ct*nks + ks)*32 + lane)*16 + e] = W[ct*16 + (lane&15)][ks*32 + (lane>>4)*16 + e]
__global__ __launch_bounds__(256) void pack_w_kernel(const float* __restrict__ W,
                                                     __bf16* __restrict__ out,
                                                     int cols, int nks)
{
  int idx = blockIdx.x * 256 + threadIdx.x;
  int total = CTILES * nks * 512;
  if (idx >= total) return;
  int e    = idx & 15;
  int lane = (idx >> 4) & 31;
  int tile = idx >> 9;
  int ks = tile % nks, ct = tile / nks;
  int wr = ct * 16 + (lane & 15);
  int wc = ks * 32 + (lane >> 4) * 16 + e;
  float v = (wr < MEM && wc < cols) ? W[wr * cols + wc] : 0.0f;
  out[idx] = (__bf16)v;
}

// ---------------- Kernel 1: embed gather + 4 input-side projections ----------------
__global__ __launch_bounds__(256) void xproj_kernel(
    const int* __restrict__ tokens, const float* __restrict__ emb,
    const __bf16* __restrict__ wxp,
    const float* __restrict__ bix, const float* __restrict__ bfx,
    const float* __restrict__ box_, const float* __restrict__ bux,
    float* __restrict__ ixw, float* __restrict__ fxw,
    float* __restrict__ oxw, float* __restrict__ uxw)
{
  __shared__ __align__(16) __bf16 xlds[64 * KPAD_X];  // 40960 B

  const int tid = threadIdx.x;
  const int node0 = blockIdx.x * 64;

  // Stage 64-node embedding tile: float4 -> 4 bf16 chunks (IN_DIM=300 is 4-aligned).
  for (int ch = tid; ch < 64 * (KPAD_X / 4); ch += 256) {
    int m = ch / (KPAD_X / 4), k4 = ch - m * (KPAD_X / 4);
    int node = node0 + m;
    int k = k4 * 4;
    float4 v = make_float4(0.f, 0.f, 0.f, 0.f);
    if (node < NODES && k < IN_DIM)
      v = *(const float4*)(emb + (size_t)tokens[node] * IN_DIM + k);
    H4 t;
    t.h[0] = (__bf16)v.x; t.h[1] = (__bf16)v.y;
    t.h[2] = (__bf16)v.z; t.h[3] = (__bf16)v.w;
    *(v2u*)(xlds + m * KPAD_X + k) = t.u;
  }
  __syncthreads();

  const int lane = tid & 31, wave = tid >> 5;
  const int hi = lane >> 4, nr = lane & 15;
  const int rt = wave & 3;   // this wave's fixed row tile

  // Hoist all 10 A fragments for this wave's row tile into VGPRs.
  v16bf afr[NKS_X];
  const __bf16* arow = xlds + (rt * 16 + nr) * KPAD_X;
#pragma unroll
  for (int ks = 0; ks < NKS_X; ++ks) afr[ks] = load_a(arow, ks * 32, hi);

  // 160 output tiles: 4 row-tiles x (4 gates x 10 col-tiles); wave w -> rt = w&3.
  for (int t = wave; t < 160; t += 8) {
    int gct = t >> 2;
    int g = gct / CTILES;
    int ct = gct - g * CTILES;
    const float* bb; float* og;
    if      (g == 0) { bb = bix;  og = ixw; }
    else if (g == 1) { bb = bfx;  og = fxw; }
    else if (g == 2) { bb = box_; og = oxw; }
    else             { bb = bux;  og = uxw; }
    const __bf16* Bp = wxp + (size_t)g * XW_SZ;

    v8f acc = {0.0f, 0.0f, 0.0f, 0.0f, 0.0f, 0.0f, 0.0f, 0.0f};
#pragma unroll
    for (int ks = 0; ks < NKS_X; ++ks) {
      v16bf b = load_bg(Bp + ((size_t)(ct * NKS_X + ks) * 32 + lane) * 16);
      acc = __builtin_amdgcn_wmma_f32_16x16x32_bf16(false, afr[ks], false, b,
                                                    (short)0, acc, false, false);
    }
    int col = ct * 16 + nr;
    float bv = (col < MEM) ? bb[col] : 0.0f;
#pragma unroll
    for (int r = 0; r < 8; ++r) {
      int node = node0 + rt * 16 + hi * 8 + r;   // C/D: M = r + 8*hi, N = lane&15
      if (node < NODES && col < MEM)
        og[(size_t)node * MEM + col] = acc[r] + bv;
    }
  }
}

// ---------------- Kernel 2: leaves (h_sum = 0, fc_sum = 0) ----------------
__global__ __launch_bounds__(256) void leaf_kernel(
    const float* __restrict__ ixw, const float* __restrict__ oxw, const float* __restrict__ uxw,
    const float* __restrict__ bih, const float* __restrict__ boh, const float* __restrict__ buh,
    float* __restrict__ cout, __bf16* __restrict__ hout)
{
  long long idx = (long long)blockIdx.x * 256 + threadIdx.x;
  long long total = (long long)LEAF_CNT * HPAD;
  if (idx >= total) return;
  int col  = (int)(idx % HPAD);
  int node = LEAF_BASE + (int)(idx / HPAD);
  if (col < MEM) {
    size_t off = (size_t)node * MEM + col;
    float iv = sigm(ixw[off] + bih[col]);
    float ov = sigm(oxw[off] + boh[col]);
    float uv = tanh_f(uxw[off] + buh[col]);
    float cv = iv * uv;
    cout[off] = cv;
    hout[(size_t)node * HPAD + col] = (__bf16)(ov * tanh_f(cv));
  } else {
    hout[(size_t)node * HPAD + col] = (__bf16)0.0f;  // keep pad region zero
  }
}

// ---------------- Kernel 3: one tree level (16 parents / 128 children per block) ----------------
__global__ __launch_bounds__(256) void level_kernel(
    const __bf16* hw, const float* cprev,
    const float* __restrict__ ixw, const float* __restrict__ fxw,
    const float* __restrict__ oxw, const float* __restrict__ uxw,
    const __bf16* __restrict__ whp,
    const float* __restrict__ bih, const float* __restrict__ bfh,
    const float* __restrict__ boh, const float* __restrict__ buh,
    float* cout, __bf16* hout,
    int base, int P)
{
  __shared__ __align__(16) __bf16 hlds[128 * MPAD];  // 40960 B  child h (bf16, padded)
  __shared__ __align__(16) float  fcl[16 * MPAD];    // 10240 B  sum_k f_k * c_k
  __shared__ __align__(16) __bf16 hsum[16 * MPAD];   //  5120 B  child-sum h

  const int tid = threadIdx.x;
  const int lane = tid & 31, wave = tid >> 5;
  const int hi = lane >> 4, nr = lane & 15;
  const int p0 = blockIdx.x * 16;
  const int childBase = base * 8 + 1;
  const int nChild = P * 8;

  const __bf16* Wihp = whp;
  const __bf16* Wfhp = whp + HW_SZ;
  const __bf16* Wohp = whp + 2 * HW_SZ;
  const __bf16* Wuhp = whp + 3 * HW_SZ;

  // Stage 128 child-h rows as aligned 16B copies (hw rows are HPAD-strided, pad pre-zeroed).
  for (int ch = tid; ch < 128 * (HPAD / 8); ch += 256) {
    int r = ch / (HPAD / 8), c8 = (ch - r * (HPAD / 8)) * 8;
    int crow = p0 * 8 + r;
    v4u v = {0u, 0u, 0u, 0u};
    if (crow < nChild) v = *(const v4u*)(hw + (size_t)(childBase + crow) * HPAD + c8);
    *(v4u*)(hlds + r * MPAD + c8) = v;
  }
  __syncthreads();

  // Child-sum h per parent, 4 columns at a time (fp32 accumulate -> bf16 operand).
  for (int i = tid; i < 16 * (MPAD / 4); i += 256) {
    int pl = i / (MPAD / 4), c4 = (i - pl * (MPAD / 4)) * 4;
    float s0 = 0.f, s1 = 0.f, s2 = 0.f, s3 = 0.f;
#pragma unroll
    for (int k = 0; k < 8; ++k) {
      H4 t; t.u = *(const v2u*)(hlds + (pl * 8 + k) * MPAD + c4);
      s0 += (float)t.h[0]; s1 += (float)t.h[1];
      s2 += (float)t.h[2]; s3 += (float)t.h[3];
    }
    H4 o;
    o.h[0] = (__bf16)s0; o.h[1] = (__bf16)s1;
    o.h[2] = (__bf16)s2; o.h[3] = (__bf16)s3;
    *(v2u*)(hsum + pl * MPAD + c4) = o.u;
  }

  // Forget-gate GEMM over 128 child rows; wave w owns row tile w. A frags hoisted.
  {
    const __bf16* arow = hlds + (wave * 16 + nr) * MPAD;
    v16bf af[NKS_H];
#pragma unroll
    for (int ks = 0; ks < NKS_H; ++ks) af[ks] = load_a(arow, ks * 32, hi);

    for (int ct = 0; ct < CTILES; ++ct) {
      v8f acc = {0.0f, 0.0f, 0.0f, 0.0f, 0.0f, 0.0f, 0.0f, 0.0f};
#pragma unroll
      for (int ks = 0; ks < NKS_H; ++ks) {
        v16bf b = load_bg(Wfhp + ((size_t)(ct * NKS_H + ks) * 32 + lane) * 16);
        acc = __builtin_amdgcn_wmma_f32_16x16x32_bf16(false, af[ks], false, b,
                                                      (short)0, acc, false, false);
      }
      // Lane's 8 accumulator rows share one parent -> local reduce, single LDS store.
      int col = ct * 16 + nr;
      int pl = wave * 2 + hi;
      if (col < MEM && (p0 + pl) < P) {
        int pg = base + p0 + pl;
        float fx = fxw[(size_t)pg * MEM + col];
        float bb = bfh[col];
        float s = 0.0f;
#pragma unroll
        for (int r = 0; r < 8; ++r) {
          int crow = p0 * 8 + wave * 16 + hi * 8 + r;
          float fv = sigm(acc[r] + bb + fx);
          s += fv * cprev[(size_t)(childBase + crow) * MEM + col];
        }
        fcl[pl * MPAD + col] = s;
      }
    }
  }
  __syncthreads();

  // i/o/u GEMMs over the 16 parent rows; waves split column tiles. A frags hoisted.
  const __bf16* srow = hsum + nr * MPAD;
  v16bf sf[NKS_H];
#pragma unroll
  for (int ks = 0; ks < NKS_H; ++ks) sf[ks] = load_a(srow, ks * 32, hi);

  for (int ct = wave; ct < CTILES; ct += 8) {
    v8f ai = {0.0f, 0.0f, 0.0f, 0.0f, 0.0f, 0.0f, 0.0f, 0.0f};
    v8f ao = ai, au = ai;
#pragma unroll
    for (int ks = 0; ks < NKS_H; ++ks) {
      size_t boff = ((size_t)(ct * NKS_H + ks) * 32 + lane) * 16;
      ai = __builtin_amdgcn_wmma_f32_16x16x32_bf16(false, sf[ks], false, load_bg(Wihp + boff),
                                                   (short)0, ai, false, false);
      ao = __builtin_amdgcn_wmma_f32_16x16x32_bf16(false, sf[ks], false, load_bg(Wohp + boff),
                                                   (short)0, ao, false, false);
      au = __builtin_amdgcn_wmma_f32_16x16x32_bf16(false, sf[ks], false, load_bg(Wuhp + boff),
                                                   (short)0, au, false, false);
    }
    int col = ct * 16 + nr;
#pragma unroll
    for (int r = 0; r < 8; ++r) {
      int pl = r + hi * 8;
      if ((p0 + pl) < P) {
        int pg = base + p0 + pl;
        if (col < MEM) {
          size_t off = (size_t)pg * MEM + col;
          float iv = sigm(ai[r] + bih[col] + ixw[off]);
          float ov = sigm(ao[r] + boh[col] + oxw[off]);
          float uv = tanh_f(au[r] + buh[col] + uxw[off]);
          float cn = iv * uv + fcl[pl * MPAD + col];
          cout[off] = cn;
          hout[(size_t)pg * HPAD + col] = (__bf16)(ov * tanh_f(cn));
        } else {
          hout[(size_t)pg * HPAD + col] = (__bf16)0.0f;  // keep pad zero
        }
      }
    }
  }
}

extern "C" void kernel_launch(void* const* d_in, const int* in_sizes, int n_in,
                              void* d_out, int out_size, void* d_ws, size_t ws_size,
                              hipStream_t stream) {
  (void)in_sizes; (void)n_in; (void)out_size; (void)ws_size;
  const int*   tokens = (const int*)d_in[0];
  const float* emb  = (const float*)d_in[4];
  const float* Wix  = (const float*)d_in[5];  const float* bix  = (const float*)d_in[6];
  const float* Wih  = (const float*)d_in[7];  const float* bih  = (const float*)d_in[8];
  const float* Wfx  = (const float*)d_in[9];  const float* bfx  = (const float*)d_in[10];
  const float* Wfh  = (const float*)d_in[11]; const float* bfh  = (const float*)d_in[12];
  const float* Wox  = (const float*)d_in[13]; const float* box_ = (const float*)d_in[14];
  const float* Woh  = (const float*)d_in[15]; const float* boh  = (const float*)d_in[16];
  const float* Wux  = (const float*)d_in[17]; const float* bux  = (const float*)d_in[18];
  const float* Wuh  = (const float*)d_in[19]; const float* buh  = (const float*)d_in[20];

  float* cout = (float*)d_out;
  // Workspace: 4 x [N,150] fp32 gate preactivations | [N,HPAD] bf16 h | packed weights.
  float* ixw = (float*)d_ws;
  float* fxw = ixw + (size_t)NODES * MEM;
  float* oxw = fxw + (size_t)NODES * MEM;
  float* uxw = oxw + (size_t)NODES * MEM;
  __bf16* hw  = (__bf16*)(uxw + (size_t)NODES * MEM);
  __bf16* wxp = hw + (size_t)NODES * HPAD;       // 4 * XW_SZ
  __bf16* whp = wxp + 4 * (size_t)XW_SZ;         // 4 * HW_SZ (order: i, f, o, u)

  dim3 blk(256);

  // Pack weights into fragment order (tiny, L2-resident afterwards).
  pack_w_kernel<<<(CTILES * NKS_X * 512 + 255) / 256, blk, 0, stream>>>(Wix, wxp + 0 * XW_SZ, IN_DIM, NKS_X);
  pack_w_kernel<<<(CTILES * NKS_X * 512 + 255) / 256, blk, 0, stream>>>(Wfx, wxp + 1 * (size_t)XW_SZ, IN_DIM, NKS_X);
  pack_w_kernel<<<(CTILES * NKS_X * 512 + 255) / 256, blk, 0, stream>>>(Wox, wxp + 2 * (size_t)XW_SZ, IN_DIM, NKS_X);
  pack_w_kernel<<<(CTILES * NKS_X * 512 + 255) / 256, blk, 0, stream>>>(Wux, wxp + 3 * (size_t)XW_SZ, IN_DIM, NKS_X);
  pack_w_kernel<<<(CTILES * NKS_H * 512 + 255) / 256, blk, 0, stream>>>(Wih, whp + 0 * HW_SZ, MEM, NKS_H);
  pack_w_kernel<<<(CTILES * NKS_H * 512 + 255) / 256, blk, 0, stream>>>(Wfh, whp + 1 * (size_t)HW_SZ, MEM, NKS_H);
  pack_w_kernel<<<(CTILES * NKS_H * 512 + 255) / 256, blk, 0, stream>>>(Woh, whp + 2 * (size_t)HW_SZ, MEM, NKS_H);
  pack_w_kernel<<<(CTILES * NKS_H * 512 + 255) / 256, blk, 0, stream>>>(Wuh, whp + 3 * (size_t)HW_SZ, MEM, NKS_H);

  xproj_kernel<<<(NODES + 63) / 64, blk, 0, stream>>>(
      tokens, emb, wxp, bix, bfx, box_, bux, ixw, fxw, oxw, uxw);

  long long leafTotal = (long long)LEAF_CNT * HPAD;
  leaf_kernel<<<(unsigned)((leafTotal + 255) / 256), blk, 0, stream>>>(
      ixw, oxw, uxw, bih, boh, buh, cout, hw);

  int P = 32768, base = 4681;  // level 5: 8^5 parents, base (8^5-1)/7
  for (int d = 5; d >= 0; --d) {
    level_kernel<<<(P + 15) / 16, blk, 0, stream>>>(
        hw, cout, ixw, fxw, oxw, uxw, whp,
        bih, bfh, boh, buh, cout, hw, base, P);
    P >>= 3;
    base = (base - 1) >> 3;
  }
}
